// sde_16509854285995
// MI455X (gfx1250) — compile-verified, matrix operand order checked by reading, not run on hardware
//
#include <hip/hip_runtime.h>
#include <hip/hip_bf16.h>

typedef __attribute__((ext_vector_type(16))) _Float16 v16h;
typedef __attribute__((ext_vector_type(8)))  _Float16 v8h;
typedef __attribute__((ext_vector_type(8)))  float    v8f;

#define BATCH   4096
#define DIM     128
#define NSTEPS  50
#define BT      64          // batch rows per workgroup
#define NP      16384       // permuted tri columns (128*128)

// ---------------------------------------------------------------------------
// Fragment loaders for v_wmma_f32_16x16x32_f16.
// Per-lane layout: lane L holds row (L&15); element e maps to
// K = k0 + (e<8 ? e+8*hi : 16+(e-8)+8*hi), hi = L>>4 => two contiguous
// 8-half (16B) runs at k0+8*hi and k0+16+8*hi.
// ---------------------------------------------------------------------------
static __device__ __forceinline__ v16h frag_lds(const _Float16 (&base)[BT * 128],
                                                int row0, int k0, int lane) {
  int r   = row0 + (lane & 15);
  int off = ((lane >> 4) & 1) * 8;
  union { v16h v; v8h h[2]; } u;
  u.h[0] = *(const v8h*)&base[r * 128 + k0 + off];
  u.h[1] = *(const v8h*)&base[r * 128 + k0 + 16 + off];
  return u.v;
}

static __device__ __forceinline__ v16h frag_glb(const _Float16* __restrict__ base,
                                                int row0, int ld, int k0,
                                                int lane) {
  int r   = row0 + (lane & 15);
  int off = ((lane >> 4) & 1) * 8;
  union { v16h v; v8h h[2]; } u;
  u.h[0] = *(const v8h*)(base + r * ld + k0 + off);
  u.h[1] = *(const v8h*)(base + r * ld + k0 + 16 + off);
  return u.v;
}

static __device__ __forceinline__ v8f wmma_f16(v16h a, v16h b, v8f c) {
  return __builtin_amdgcn_wmma_f32_16x16x32_f16(false, a, false, b,
                                                (short)0, c, false, false);
}

// Branchless tanh: (e-1)*rcp(e+1), e = exp(2x). v_exp_f32 + v_rcp_f32.
static __device__ __forceinline__ float fast_tanh(float x) {
  float cx = fminf(fmaxf(x, -15.f), 15.f);
  float e  = __expf(2.f * cx);
  return (e - 1.f) * __builtin_amdgcn_rcpf(e + 1.f);
}

// ---------------------------------------------------------------------------
// 64x128 GEMM layer: out = act(A[64x128] @ Wt^T + bias) (+ optional skip)
// A/out/skip in LDS (f16, array refs), Wt in global (f16, [128 n][128 k]).
// Wave w owns a fixed 16-row block (row0 = (w>>1)*16) and 4 N-tiles; the A
// fragment per k-step is loaded once and reused by 4 wmma (k0-outer order).
// MODE: 0 = tanh, 1 = relu. SKIP adds skip[] after activation.
// ---------------------------------------------------------------------------
template <int MODE, bool SKIP>
static __device__ __forceinline__ void gemm_layer(
    const _Float16 (&A)[BT * 128], const _Float16* __restrict__ Wt,
    const float* __restrict__ bias, _Float16 (&out)[BT * 128],
    const _Float16 (&skip)[BT * 128], int lane, int w) {
  int row0  = (w >> 1) * 16;
  int nbase = (w & 1) * 64;
  v8f c[4] = {{}, {}, {}, {}};
#pragma unroll
  for (int k0 = 0; k0 < 128; k0 += 32) {
    v16h a = frag_lds(A, row0, k0, lane);
#pragma unroll
    for (int q = 0; q < 4; ++q) {
      v16h b = frag_glb(Wt, nbase + q * 16, 128, k0, lane);
      c[q] = wmma_f16(a, b, c[q]);
    }
  }
  int hi = (lane >> 4) & 1;
#pragma unroll
  for (int q = 0; q < 4; ++q) {
    int N = nbase + q * 16 + (lane & 15);
    float bv = bias[N];
#pragma unroll
    for (int r = 0; r < 8; ++r) {
      int M = row0 + r + 8 * hi;
      float v = c[q][r] + bv;
      v = (MODE == 0) ? fast_tanh(v) : fmaxf(v, 0.f);
      if (SKIP) v += (float)skip[M * 128 + N];
      out[M * 128 + N] = (_Float16)v;
    }
  }
}

// Final drift layer: writes f32 to global (no activation).
static __device__ __forceinline__ void gemm_out(
    const _Float16 (&A)[BT * 128], const _Float16* __restrict__ Wt,
    const float* __restrict__ bias, float* __restrict__ gout, int b0,
    int lane, int w) {
  int row0  = (w >> 1) * 16;
  int nbase = (w & 1) * 64;
  v8f c[4] = {{}, {}, {}, {}};
#pragma unroll
  for (int k0 = 0; k0 < 128; k0 += 32) {
    v16h a = frag_lds(A, row0, k0, lane);
#pragma unroll
    for (int q = 0; q < 4; ++q) {
      v16h b = frag_glb(Wt, nbase + q * 16, 128, k0, lane);
      c[q] = wmma_f16(a, b, c[q]);
    }
  }
  int hi = (lane >> 4) & 1;
#pragma unroll
  for (int q = 0; q < 4; ++q) {
    int N = nbase + q * 16 + (lane & 15);
    float bv = bias[N];
#pragma unroll
    for (int r = 0; r < 8; ++r) {
      int M = row0 + r + 8 * hi;
      gout[(b0 + M) * 128 + N] = c[q][r] + bv;
    }
  }
}

// ---------------------------------------------------------------------------
// Kernel 1: both MLP hidden chains + drift output for one timestep.
// ---------------------------------------------------------------------------
__global__ __launch_bounds__(256) void mlp_kernel(
    const float* __restrict__ state,
    const _Float16* Wd1, const _Float16* Wd2, const _Float16* Wd3,
    const _Float16* Wdo,
    const _Float16* Wf1, const _Float16* Wf2, const _Float16* Wf3,
    const float* bd1, const float* bd2, const float* bd3, const float* bdo,
    const float* bf1, const float* bf2, const float* bf3,
    float* __restrict__ drift_out, _Float16* __restrict__ h3diff) {
  __shared__ __align__(16) _Float16 xbuf[BT * 128];
  __shared__ __align__(16) _Float16 hA[BT * 128];
  __shared__ __align__(16) _Float16 hB[BT * 128];

  int b0   = blockIdx.x * BT;
  int tid  = threadIdx.x;
  int lane = tid & 31;
  int w    = tid >> 5;

  for (int e = tid; e < BT * 128; e += 256)
    xbuf[e] = (_Float16)state[b0 * 128 + e];
  __syncthreads();

  // ---- drift MLP ----
  gemm_layer<0, false>(xbuf, Wd1, bd1, hA, hA, lane, w); __syncthreads();
  gemm_layer<0, true >(hA,   Wd2, bd2, hB, hA, lane, w); __syncthreads(); // h2+h1
  gemm_layer<1, false>(hB,   Wd3, bd3, hA, hA, lane, w); __syncthreads();
  gemm_out(hA, Wdo, bdo, drift_out, b0, lane, w);        __syncthreads();

  // ---- diff MLP hidden chain ----
  gemm_layer<0, false>(xbuf, Wf1, bf1, hA, hA, lane, w); __syncthreads();
  gemm_layer<0, true >(hA,   Wf2, bf2, hB, hA, lane, w); __syncthreads();
  gemm_layer<1, false>(hB,   Wf3, bf3, hA, hA, lane, w); __syncthreads();

  for (int e = tid; e < BT * 128; e += 256)
    h3diff[b0 * 128 + e] = hA[e];
}

// ---------------------------------------------------------------------------
// Kernel 2: fused triangular GEMM (h3diff @ Wtri), L@z, diag exp/log-det,
// Euler step, Huber loss accumulation.
// A-fragments (16x128 slab per wave) are hoisted into registers for the whole
// kernel; only B (Wtri, L2-resident) streams from global.
// ---------------------------------------------------------------------------
__global__ __launch_bounds__(256) void tri_kernel(
    const _Float16* __restrict__ h3diff,
    const _Float16* __restrict__ Wtri,   // [16384 p][128 k]
    const float* __restrict__ btri,      // [16384]
    const float* __restrict__ zstep,     // [4096][128]
    const float* __restrict__ drift_out,
    const float* __restrict__ yhat_step,
    float* __restrict__ state,
    float* __restrict__ huber_acc, float* __restrict__ ld_acc, int step) {
  __shared__ __align__(16) _Float16 abuf[BT * 128];   // h3 tile
  __shared__ __align__(16) _Float16 zbuf[BT * 128];   // z tile (f16)
  __shared__ float Lz[BT * 128];                      // L@z result
  __shared__ float red[256 + BT];                     // huber tree + logdet[64]

  int b0   = blockIdx.x * BT;
  int tid  = threadIdx.x;
  int lane = tid & 31;
  int w    = tid >> 5;

  for (int e = tid; e < BT * 128; e += 256) {
    abuf[e] = h3diff[b0 * 128 + e];
    zbuf[e] = (_Float16)zstep[b0 * 128 + e];
  }
  if (tid < BT) red[256 + tid] = 0.f;
  __syncthreads();

  int mt   = w & 3;        // batch sub-tile (16 rows)
  int par  = w >> 2;       // i parity
  int row0 = mt * 16;
  int N    = lane & 15;
  int hi   = (lane >> 4) & 1;

  // Hoist the wave's entire A slab (16 rows x 128 K) into registers.
  v16h afr[4];
#pragma unroll
  for (int kk = 0; kk < 4; ++kk) afr[kk] = frag_lds(abuf, row0, kk * 32, lane);

  for (int i = par; i < 128; i += 2) {
    float acc[8];
#pragma unroll
    for (int r = 0; r < 8; ++r) acc[r] = 0.f;

    for (int j0 = 0; j0 <= i; j0 += 16) {
      int p0 = i * 128 + j0;
      v8f c = {};
#pragma unroll
      for (int kk = 0; kk < 4; ++kk) {
        v16h b = frag_glb(Wtri, p0, 128, kk * 32, lane);
        c = wmma_f16(afr[kk], b, c);
      }
      int j = j0 + N;
      float bv   = btri[p0 + N];
      bool  diag = (j == i);
#pragma unroll
      for (int r = 0; r < 8; ++r) {
        int M = row0 + r + 8 * hi;
        float v  = c[r] + bv;                 // exactly 0 above the diagonal
        float zv = (float)zbuf[M * 128 + j];
        if (diag) {
          atomicAdd(&red[256 + M], v);        // log(det) = sum of raw diag v
          acc[r] += __expf(v) * zv;
        } else {
          acc[r] += v * zv;
        }
      }
    }
    // one cross-lane (j) reduction per output row i
#pragma unroll
    for (int r = 0; r < 8; ++r) {
      float t = acc[r];
#pragma unroll
      for (int s = 1; s < 16; s <<= 1) t += __shfl_xor(t, s, 32);
      if (N == 0) Lz[(row0 + r + 8 * hi) * 128 + i] = t;  // unique writer
    }
  }
  __syncthreads();

  // Euler-Maruyama step + Huber
  float hsum = 0.f;
  for (int e = tid; e < BT * 128; e += 256) {
    int g = b0 * 128 + e;
    float out = state[g] + drift_out[g] + Lz[e];
    float err = out - yhat_step[g];
    float ae  = fabsf(err);
    hsum += (ae <= 0.5f) ? 0.5f * err * err : 0.5f * ae - 0.125f;
    state[g] = out;
  }
  red[tid] = hsum;
  __syncthreads();
  for (int s = 128; s > 0; s >>= 1) {
    if (tid < s) red[tid] += red[tid + s];
    __syncthreads();
  }
  if (tid == 0) {
    atomicAdd(&huber_acc[step], red[0]);
    float l = 0.f;
    for (int q = 0; q < BT; ++q) l += red[256 + q];
    atomicAdd(&ld_acc[step], l);
  }
}

// ---------------------------------------------------------------------------
// Prep kernels
// ---------------------------------------------------------------------------
__global__ void prep_wt(_Float16* dst, const float* src, int fanin, int fanout) {
  int idx = blockIdx.x * 256 + threadIdx.x;
  if (idx >= 128 * 128) return;
  int n = idx >> 7, k = idx & 127;
  float v = (n < fanout && k < fanin) ? src[k * fanout + n] : 0.f;
  dst[idx] = (_Float16)v;
}

__global__ void prep_bias(float* dst, const float* src, int fanout) {
  int n = threadIdx.x;
  if (n < 128) dst[n] = (n < fanout) ? src[n] : 0.f;
}

// fill_triangular-permuted, transposed, padded diff_Wo: [p=i*128+j][k]
__global__ void prep_wtri(_Float16* dst, const float* wo) {
  int idx = blockIdx.x * 256 + threadIdx.x;
  if (idx >= NP * 128) return;
  int p = idx >> 7, k = idx & 127;
  int i = p >> 7, j = p & 127;
  float v = 0.f;
  if (j <= i && k < 100) {
    int t = i * 128 + j;
    int m = (t < 8128) ? t + 128 : 16383 - t;
    v = wo[k * 8256 + m];
  }
  dst[idx] = (_Float16)v;
}

__global__ void prep_btri(float* dst, const float* bo) {
  int p = blockIdx.x * 256 + threadIdx.x;
  if (p >= NP) return;
  int i = p >> 7, j = p & 127;
  float v = 0.f;
  if (j <= i) {
    int t = i * 128 + j;
    int m = (t < 8128) ? t + 128 : 16383 - t;
    v = bo[m];
  }
  dst[p] = v;
}

__global__ void prep_state(float* state, const float* init, float* accs) {
  int idx = blockIdx.x * 256 + threadIdx.x;
  if (idx < BATCH * DIM) state[idx] = init[idx];
  if (idx < 128) accs[idx] = 0.f;   // huber_acc[50] + ld_acc[50]
}

__global__ void finalize(const float* hacc, const float* ldacc, float* out) {
  if (blockIdx.x == 0 && threadIdx.x == 0) {
    const float sw = 1.0f / (float)NSTEPS;
    float run = 0.f;
    for (int s = NSTEPS - 1; s >= 0; --s) {
      float li = hacc[s] * (sw / (float)(BATCH * DIM))
               - ldacc[s] * (sw * 1.0e-4f / (float)BATCH);
      run += li;
      out[s] = run;
    }
  }
}

// ---------------------------------------------------------------------------
extern "C" void kernel_launch(void* const* d_in, const int* in_sizes, int n_in,
                              void* d_out, int out_size, void* d_ws,
                              size_t ws_size, hipStream_t stream) {
  const float* init_state = (const float*)d_in[0];
  const float* yhat       = (const float*)d_in[1];
  const float* z          = (const float*)d_in[2];
  const float* dW[4] = {(const float*)d_in[5],  (const float*)d_in[7],
                        (const float*)d_in[9],  (const float*)d_in[11]};
  const float* db[4] = {(const float*)d_in[6],  (const float*)d_in[8],
                        (const float*)d_in[10], (const float*)d_in[12]};
  const float* fW[4] = {(const float*)d_in[13], (const float*)d_in[15],
                        (const float*)d_in[17], (const float*)d_in[19]};
  const float* fb[4] = {(const float*)d_in[14], (const float*)d_in[16],
                        (const float*)d_in[18], (const float*)d_in[20]};

  char* ws = (char*)d_ws;
  size_t o = 0;
  auto alloc = [&](size_t bytes) -> void* {
    void* p = ws + o;
    o = (o + bytes + 255) & ~(size_t)255;
    return p;
  };

  _Float16* Wd[4]; for (int l = 0; l < 4; ++l) Wd[l] = (_Float16*)alloc(128 * 128 * 2);
  _Float16* Wf[3]; for (int l = 0; l < 3; ++l) Wf[l] = (_Float16*)alloc(128 * 128 * 2);
  _Float16* Wtri  = (_Float16*)alloc((size_t)NP * 128 * 2);
  float*    btri  = (float*)alloc(NP * 4);
  float*    bdp[4]; for (int l = 0; l < 4; ++l) bdp[l] = (float*)alloc(128 * 4);
  float*    bfp[3]; for (int l = 0; l < 3; ++l) bfp[l] = (float*)alloc(128 * 4);
  float*    state     = (float*)alloc((size_t)BATCH * DIM * 4);
  float*    drift_out = (float*)alloc((size_t)BATCH * DIM * 4);
  _Float16* h3diff    = (_Float16*)alloc((size_t)BATCH * DIM * 2);
  float*    huber_acc = (float*)alloc(NSTEPS * 4);
  float*    ld_acc    = (float*)alloc(NSTEPS * 4);

  // --- prep: transposed/padded f16 weights, permuted tri weights ---
  const int fin_d[4] = {128, 100, 100, 100}, fout_d[4] = {100, 100, 100, 128};
  for (int l = 0; l < 4; ++l) {
    prep_wt<<<64, 256, 0, stream>>>(Wd[l], dW[l], fin_d[l], fout_d[l]);
    prep_bias<<<1, 128, 0, stream>>>(bdp[l], db[l], fout_d[l]);
  }
  for (int l = 0; l < 3; ++l) {
    prep_wt<<<64, 256, 0, stream>>>(Wf[l], fW[l], fin_d[l], fout_d[l]);
    prep_bias<<<1, 128, 0, stream>>>(bfp[l], fb[l], fout_d[l]);
  }
  prep_wtri<<<(NP * 128 + 255) / 256, 256, 0, stream>>>(Wtri, fW[3]);
  prep_btri<<<(NP + 255) / 256, 256, 0, stream>>>(btri, fb[3]);
  prep_state<<<(BATCH * DIM + 255) / 256, 256, 0, stream>>>(state, init_state,
                                                            huber_acc);

  // --- 50 sequential Euler-Maruyama steps ---
  const int nblk = BATCH / BT;   // 64 workgroups
  for (int s = 0; s < NSTEPS; ++s) {
    mlp_kernel<<<nblk, 256, 0, stream>>>(
        state, Wd[0], Wd[1], Wd[2], Wd[3], Wf[0], Wf[1], Wf[2],
        bdp[0], bdp[1], bdp[2], bdp[3], bfp[0], bfp[1], bfp[2],
        drift_out, h3diff);
    tri_kernel<<<nblk, 256, 0, stream>>>(
        h3diff, Wtri, btri, z + (size_t)s * BATCH * DIM, drift_out,
        yhat + (size_t)s * BATCH * DIM, state, huber_acc, ld_acc, s);
  }

  finalize<<<1, 64, 0, stream>>>(huber_acc, ld_acc, (float*)d_out);
}
